// MultiHeadAttention_15092515078557
// MI455X (gfx1250) — matrix operand order repaired and checked by reading, not compiled, not run on hardware
//
#include <hip/hip_runtime.h>

typedef __bf16 bf16_t;
typedef __attribute__((ext_vector_type(16))) __bf16 v16bf;
typedef __attribute__((ext_vector_type(8)))  __bf16 v8bf;
typedef __attribute__((ext_vector_type(4)))  __bf16 v4bf;
typedef __attribute__((ext_vector_type(8)))  float  v8f;
typedef __attribute__((ext_vector_type(4)))  unsigned uint32x4;
typedef __attribute__((ext_vector_type(4)))  int      int32x4;
typedef __attribute__((ext_vector_type(8)))  int      int32x8;

// Problem constants (fixed by the reference harness)
constexpr int Bn = 4;
constexpr int Tn = 2048;
constexpr int Dn = 1024;
constexpr int Hn = 16;
constexpr int HD = 64;
constexpr int Mn = Bn * Tn;   // 8192 rows

// ---------------------------------------------------------------------------
// helpers
// ---------------------------------------------------------------------------
__device__ __forceinline__ bf16_t f32_to_bf16(float f) {
  unsigned u = __builtin_bit_cast(unsigned, f);
  u += 0x7FFFu + ((u >> 16) & 1u);            // round-to-nearest-even
  unsigned short s = (unsigned short)(u >> 16);
  return __builtin_bit_cast(bf16_t, s);
}

// A-matrix 16x32 bf16 fragment: lane (0..15|16..31) holds row (lane&15),
// K = {2*kb..2*kb+15} (contiguous) where kb = (lane>>4)*8.
__device__ __forceinline__ v16bf load_afrag(const bf16_t* row, int kb) {
  const v8bf lo = *(const v8bf*)(row + 2 * kb);
  const v8bf hi = *(const v8bf*)(row + 2 * kb + 8);
  v16bf r;
#pragma unroll
  for (int i = 0; i < 8; ++i) { r[i] = lo[i]; r[i + 8] = hi[i]; }
  return r;
}

// B-matrix 32x16 bf16 fragment: lane holds column (lane&15) and 16 contiguous
// K values starting at (lane>>4)*16.  `p` points at those 16 values.
__device__ __forceinline__ v16bf load_bfrag(const bf16_t* p) {
  const v8bf lo = *(const v8bf*)(p);
  const v8bf hi = *(const v8bf*)(p + 8);
  v16bf r;
#pragma unroll
  for (int i = 0; i < 8; ++i) { r[i] = lo[i]; r[i + 8] = hi[i]; }
  return r;
}

__device__ __forceinline__ v8f wmma_bf16(v16bf a, v16bf b, v8f c) {
  return __builtin_amdgcn_wmma_f32_16x16x32_bf16(false, a, false, b,
                                                 (short)0, c, false, false);
}

// ---------------------------------------------------------------------------
// kernel 1: fp32 -> bf16 cast (x)
// ---------------------------------------------------------------------------
__global__ __launch_bounds__(256) void cast_kernel(const float* __restrict__ x,
                                                   bf16_t* __restrict__ xb, int n4) {
  int i = blockIdx.x * 256 + threadIdx.x;
  if (i < n4) {
    float4 f = ((const float4*)x)[i];
    v4bf o;
    o[0] = f32_to_bf16(f.x); o[1] = f32_to_bf16(f.y);
    o[2] = f32_to_bf16(f.z); o[3] = f32_to_bf16(f.w);
    ((v4bf*)xb)[i] = o;
  }
}

// ---------------------------------------------------------------------------
// kernel 2: transpose + cast weight  W[in][out] fp32 -> Wt[out][in] bf16
// ---------------------------------------------------------------------------
__global__ __launch_bounds__(256) void transpose_cast_kernel(const float* __restrict__ W,
                                                             bf16_t* __restrict__ Wt) {
  __shared__ float tile[32][33];
  const int tx = threadIdx.x & 31;
  const int ty = threadIdx.x >> 5;          // 0..7
  const int k0 = blockIdx.y * 32;
  const int n0 = blockIdx.x * 32;
#pragma unroll
  for (int i = 0; i < 4; ++i)
    tile[ty + i * 8][tx] = W[(size_t)(k0 + ty + i * 8) * Dn + n0 + tx];
  __syncthreads();
#pragma unroll
  for (int i = 0; i < 4; ++i)
    Wt[(size_t)(n0 + ty + i * 8) * Dn + k0 + tx] = f32_to_bf16(tile[tx][ty + i * 8]);
}

// ---------------------------------------------------------------------------
// kernel 3: WMMA GEMM  C[M,N] = A[M,K] * Bt[N,K]^T + bias[N]
// Register-blocked: each wave owns a 32(M) x 64(N) tile -> 8 WMMAs per k-step
// from 2 A-fragments + 4 B-fragments (0.75 fragment loads per WMMA).
// Block = 256 threads (8 waves as 4(M) x 2(N)), block tile = 128 x 128.
// grid = (N/128, M/128).
// mode 0: store bf16 to [B,H,T,Hd]    (Q / K projections)
// mode 1: store bf16 to [B,H,Hd,T]    (V projection, transposed for PV wmma)
// mode 2: store fp32 to [M,N]         (output projection)
// ---------------------------------------------------------------------------
__global__ __launch_bounds__(256) void gemm_kernel(const bf16_t* __restrict__ A,
                                                   const bf16_t* __restrict__ Bt,
                                                   const float* __restrict__ bias,
                                                   bf16_t* __restrict__ outb,
                                                   float* __restrict__ outf,
                                                   int mode) {
  const int lane = threadIdx.x & 31;
  const int wave = threadIdx.x >> 5;
  const int rm = blockIdx.y * 128 + (wave & 3) * 32;   // 32 rows for this wave
  const int rn = blockIdx.x * 128 + (wave >> 2) * 64;  // 64 cols for this wave

  const int arow = lane & 15;
  const int akb  = (lane >> 4) << 3;        // A fragment K base (0 or 8)
  const int bkb  = (lane >> 4) << 4;        // B fragment K base (0 or 16)

  const bf16_t* ar0 = A  + (size_t)(rm + arow) * Dn;
  const bf16_t* ar1 = A  + (size_t)(rm + 16 + arow) * Dn;
  const bf16_t* br0 = Bt + (size_t)(rn + arow) * Dn + bkb;

  v8f acc[8];
#pragma unroll
  for (int i = 0; i < 8; ++i) acc[i] = (v8f){};

#pragma unroll 2
  for (int k = 0; k < Dn; k += 32) {
    const v16bf a0 = load_afrag(ar0 + k, akb);
    const v16bf a1 = load_afrag(ar1 + k, akb);
    const v16bf b0 = load_bfrag(br0 + k);
    const v16bf b1 = load_bfrag(br0 + 16 * Dn + k);
    const v16bf b2 = load_bfrag(br0 + 32 * Dn + k);
    const v16bf b3 = load_bfrag(br0 + 48 * Dn + k);
    acc[0] = wmma_bf16(a0, b0, acc[0]);
    acc[1] = wmma_bf16(a0, b1, acc[1]);
    acc[2] = wmma_bf16(a0, b2, acc[2]);
    acc[3] = wmma_bf16(a0, b3, acc[3]);
    acc[4] = wmma_bf16(a1, b0, acc[4]);
    acc[5] = wmma_bf16(a1, b1, acc[5]);
    acc[6] = wmma_bf16(a1, b2, acc[6]);
    acc[7] = wmma_bf16(a1, b3, acc[7]);
    __builtin_prefetch(ar0 + k + 512, 0, 0);   // next K-slab of A
    __builtin_prefetch(br0 + k + 512, 0, 0);   // next K-slab of B
  }

#pragma unroll
  for (int mi = 0; mi < 2; ++mi) {
#pragma unroll
    for (int ni = 0; ni < 4; ++ni) {
      const v8f c = acc[mi * 4 + ni];
      const int n     = rn + ni * 16 + (lane & 15);
      const int mbase = rm + mi * 16 + ((lane >> 4) << 3);
      const float bv  = bias[n];
      if (mode == 2) {
#pragma unroll
        for (int r = 0; r < 8; ++r)
          outf[(size_t)(mbase + r) * Dn + n] = c[r] + bv;
      } else {
        const int h = n >> 6, hd = n & 63;
#pragma unroll
        for (int r = 0; r < 8; ++r) {
          const int m = mbase + r;
          const int bidx = m >> 11;             // m / Tn
          const int t    = m & (Tn - 1);
          size_t addr;
          if (mode == 0) addr = (((size_t)(bidx * Hn + h) * Tn) + t) * HD + hd;
          else           addr = (((size_t)(bidx * Hn + h) * HD) + hd) * Tn + t;
          outb[addr] = f32_to_bf16(c[r] + bv);
        }
      }
    }
  }
}

// ---------------------------------------------------------------------------
// kernel 4: flash attention.
// grid = (Tn/64, Bn*Hn), block = 128 (4 waves; each wave owns 16 query rows)
// Q,K: [B,H,T,Hd] bf16;  Vt: [B,H,Hd,T] bf16;  ctx out: [B*T, D] bf16
// K chunk staged via the Tensor Data Mover (TDM); V chunk via plain
// cooperative loads.
// ---------------------------------------------------------------------------
__global__ __launch_bounds__(128) void attn_kernel(const bf16_t* __restrict__ Q,
                                                   const bf16_t* __restrict__ K,
                                                   const bf16_t* __restrict__ Vt,
                                                   bf16_t* __restrict__ ctx) {
  __shared__ __align__(16) bf16_t sK[32][HD];      // 32 keys  x 64 dims
  __shared__ __align__(16) bf16_t sV[HD][32];      // 64 dims  x 32 keys
  __shared__ __align__(16) bf16_t sP[4][16][32];   // per-wave P tile

  const int tid  = threadIdx.x;
  const int lane = tid & 31;
  const int wave = tid >> 5;
  const int bh   = blockIdx.y;
  const int b    = bh >> 4;
  const int h    = bh & 15;
  const int qbase = (blockIdx.x * 4 + wave) * 16;

  const bf16_t* Qbh = Q  + (size_t)bh * Tn * HD;
  const bf16_t* Kbh = K  + (size_t)bh * Tn * HD;
  const bf16_t* Vbh = Vt + (size_t)bh * HD * Tn;

  const int arow = lane & 15;
  const int akb  = (lane >> 4) << 3;
  const int bcol = lane & 15;
  const int bkb  = (lane >> 4) << 4;

  // Q fragments for dims 0..31 and 32..63 (kept in registers for all chunks)
  const bf16_t* qrow = Qbh + (size_t)(qbase + arow) * HD;
  const v16bf aQ0 = load_afrag(qrow, akb);
  const v16bf aQ1 = load_afrag(qrow + 32, akb);

  v8f acc0 = {}, acc1 = {}, acc2 = {}, acc3 = {};
  float mrow[8], lrow[8];
#pragma unroll
  for (int r = 0; r < 8; ++r) { mrow[r] = -1e30f; lrow[r] = 0.0f; }
  const float scale = 0.125f;   // 1/sqrt(64)

  const unsigned ldsK = (unsigned)(unsigned long long)&sK[0][0];

  for (int kb0 = 0; kb0 < Tn; kb0 += 32) {
    // ---- stage K chunk via TDM: contiguous 32*64 bf16 = 4 KB ------------
    if (wave == 0) {
      const unsigned long long ga =
          (unsigned long long)(const void*)(Kbh + (size_t)kb0 * HD);
      // D# group 0: count=1, lds_addr, 57-bit global addr, type=2
      uint32x4 g0 = { 1u, ldsK, (unsigned)ga,
                      (unsigned)((ga >> 32) & 0x01FFFFFFu) | 0x80000000u };
      // D# group 1: data_size=2B; tensor_dim0=2048, tensor_dim1=1,
      // tile_dim0=2048 (one contiguous 1-D tile), dim0 stride=2048
      int32x8 g1 = { 0x00010000, 0x08000000, 0x00010000, 0x08000000,
                     0, 2048, 0, 0 };
      int32x4 gz = { 0, 0, 0, 0 };
#if defined(__clang_major__) && (__clang_major__ >= 23)
      int32x8 gz8 = { 0, 0, 0, 0, 0, 0, 0, 0 };
      __builtin_amdgcn_tensor_load_to_lds(g0, g1, gz, gz, gz8, 0);
#else
      __builtin_amdgcn_tensor_load_to_lds(g0, g1, gz, gz, 0);
#endif
      __builtin_amdgcn_s_wait_tensorcnt(0);
    }
    // ---- cooperative stage of V chunk into LDS --------------------------
    for (int i = tid; i < HD * 32 / 8; i += 128) {
      const int d  = i >> 2;
      const int c8 = i & 3;
      ((v8bf*)&sV[d][0])[c8] = *(const v8bf*)(Vbh + (size_t)d * Tn + kb0 + c8 * 8);
    }
    __syncthreads();

    // ---- S = Q * K^T (two 16-key column tiles) --------------------------
    v8f s0 = {}, s1 = {};
    {
      v16bf bk = load_bfrag(&sK[bcol][bkb]);            // keys 0..15, dims 0..31
      s0 = wmma_bf16(aQ0, bk, s0);
      bk = load_bfrag(&sK[bcol][32 + bkb]);             // dims 32..63
      s0 = wmma_bf16(aQ1, bk, s0);
      bk = load_bfrag(&sK[16 + bcol][bkb]);             // keys 16..31
      s1 = wmma_bf16(aQ0, bk, s1);
      bk = load_bfrag(&sK[16 + bcol][32 + bkb]);
      s1 = wmma_bf16(aQ1, bk, s1);
    }

    // ---- online softmax (row stats via half-wave shfl reductions) -------
    float p0[8], p1[8], alpha[8];
#pragma unroll
    for (int r = 0; r < 8; ++r) {
      const float v0 = s0[r] * scale;
      const float v1 = s1[r] * scale;
      float mx = fmaxf(v0, v1);
      mx = fmaxf(mx, __shfl_xor(mx, 1));
      mx = fmaxf(mx, __shfl_xor(mx, 2));
      mx = fmaxf(mx, __shfl_xor(mx, 4));
      mx = fmaxf(mx, __shfl_xor(mx, 8));
      const float mnew = fmaxf(mrow[r], mx);
      const float a = __expf(mrow[r] - mnew);
      p0[r] = __expf(v0 - mnew);
      p1[r] = __expf(v1 - mnew);
      float rs = p0[r] + p1[r];
      rs += __shfl_xor(rs, 1);
      rs += __shfl_xor(rs, 2);
      rs += __shfl_xor(rs, 4);
      rs += __shfl_xor(rs, 8);
      lrow[r] = lrow[r] * a + rs;
      mrow[r] = mnew;
      alpha[r] = a;
    }
#pragma unroll
    for (int r = 0; r < 8; ++r) {
      acc0[r] *= alpha[r]; acc1[r] *= alpha[r];
      acc2[r] *= alpha[r]; acc3[r] *= alpha[r];
    }

    // ---- C-fragment -> A-fragment re-layout of P via per-wave LDS tile --
    {
      const int prow = (lane >> 4) << 3;   // rows 0..7 or 8..15
      const int pcol = lane & 15;
#pragma unroll
      for (int r = 0; r < 8; ++r) {
        sP[wave][prow + r][pcol]      = f32_to_bf16(p0[r]);
        sP[wave][prow + r][pcol + 16] = f32_to_bf16(p1[r]);
      }
    }
    // same-wave LDS ops are in order; safe to read back without a barrier
    const v16bf aP = load_afrag(&sP[wave][arow][0], akb);

    // ---- ctx += P * V (four 16-dim column tiles) ------------------------
    {
      v16bf bv = load_bfrag(&sV[bcol][bkb]);
      acc0 = wmma_bf16(aP, bv, acc0);
      bv = load_bfrag(&sV[16 + bcol][bkb]);
      acc1 = wmma_bf16(aP, bv, acc1);
      bv = load_bfrag(&sV[32 + bcol][bkb]);
      acc2 = wmma_bf16(aP, bv, acc2);
      bv = load_bfrag(&sV[48 + bcol][bkb]);
      acc3 = wmma_bf16(aP, bv, acc3);
    }
    __syncthreads();   // before next chunk overwrites sK/sV
  }

  // ---- normalize and store ctx in [B*T, D] bf16 for the O projection ----
#pragma unroll
  for (int r = 0; r < 8; ++r) {
    const float il = 1.0f / lrow[r];
    const int m = qbase + ((lane >> 4) << 3) + r;
    bf16_t* crow = ctx + ((size_t)(b * Tn + m)) * Dn + h * HD + (lane & 15);
    crow[0]  = f32_to_bf16(acc0[r] * il);
    crow[16] = f32_to_bf16(acc1[r] * il);
    crow[32] = f32_to_bf16(acc2[r] * il);
    crow[48] = f32_to_bf16(acc3[r] * il);
  }
}

// ---------------------------------------------------------------------------
// launcher
// ---------------------------------------------------------------------------
extern "C" void kernel_launch(void* const* d_in, const int* in_sizes, int n_in,
                              void* d_out, int out_size, void* d_ws, size_t ws_size,
                              hipStream_t stream) {
  (void)in_sizes; (void)n_in; (void)out_size; (void)ws_size;
  const float* x  = (const float*)d_in[0];
  const float* Wq = (const float*)d_in[1];
  const float* bq = (const float*)d_in[2];
  const float* Wk = (const float*)d_in[3];
  const float* bk = (const float*)d_in[4];
  const float* Wv = (const float*)d_in[5];
  const float* bv = (const float*)d_in[6];
  const float* Wo = (const float*)d_in[7];
  const float* bo = (const float*)d_in[8];
  float* out = (float*)d_out;

  char* p = (char*)d_ws;
  bf16_t* xb  = (bf16_t*)p; p += (size_t)Mn * Dn * 2;
  bf16_t* Wtq = (bf16_t*)p; p += (size_t)Dn * Dn * 2;
  bf16_t* Wtk = (bf16_t*)p; p += (size_t)Dn * Dn * 2;
  bf16_t* Wtv = (bf16_t*)p; p += (size_t)Dn * Dn * 2;
  bf16_t* Wto = (bf16_t*)p; p += (size_t)Dn * Dn * 2;
  bf16_t* Qb  = (bf16_t*)p; p += (size_t)Mn * Dn * 2;
  bf16_t* Kb  = (bf16_t*)p; p += (size_t)Mn * Dn * 2;
  bf16_t* Vtb = (bf16_t*)p; p += (size_t)Mn * Dn * 2;
  bf16_t* ctx = xb;   // xb is dead after the V projection; reuse for ctx

  // 1) cast x to bf16
  cast_kernel<<<(Mn * Dn / 4 + 255) / 256, 256, 0, stream>>>(x, xb, Mn * Dn / 4);

  // 2) transpose-cast the four weights to [out,in] bf16
  dim3 tg(Dn / 32, Dn / 32);
  transpose_cast_kernel<<<tg, 256, 0, stream>>>(Wq, Wtq);
  transpose_cast_kernel<<<tg, 256, 0, stream>>>(Wk, Wtk);
  transpose_cast_kernel<<<tg, 256, 0, stream>>>(Wv, Wtv);
  transpose_cast_kernel<<<tg, 256, 0, stream>>>(Wo, Wto);

  // 3) Q/K/V projections (WMMA GEMMs)
  dim3 gg(Dn / 128, Mn / 128);
  gemm_kernel<<<gg, 256, 0, stream>>>(xb, Wtq, bq, Qb,  nullptr, 0);
  gemm_kernel<<<gg, 256, 0, stream>>>(xb, Wtk, bk, Kb,  nullptr, 0);
  gemm_kernel<<<gg, 256, 0, stream>>>(xb, Wtv, bv, Vtb, nullptr, 1);

  // 4) flash attention -> ctx
  attn_kernel<<<dim3(Tn / 64, Bn * Hn), 128, 0, stream>>>(Qb, Kb, Vtb, ctx);

  // 5) output projection, fp32 + bias to d_out
  gemm_kernel<<<gg, 256, 0, stream>>>(ctx, Wto, bo, nullptr, out, 2);
}
